// Head_3624952397970
// MI455X (gfx1250) — matrix operand order repaired and checked by reading, not compiled, not run on hardware
//
#include <hip/hip_runtime.h>
#include <hip/hip_bf16.h>

// ---------------------------------------------------------------------------
// Fused causal attention head for MI455X (gfx1250, wave32, WMMA bf16 + TDM)
// B=8, T=2048, E=1024, H=64
// ---------------------------------------------------------------------------

typedef __attribute__((ext_vector_type(16))) __bf16 v16bf;
typedef __attribute__((ext_vector_type(8)))  __bf16 v8bf;
typedef __attribute__((ext_vector_type(8)))  float  v8f;
typedef unsigned int u32x4 __attribute__((ext_vector_type(4)));
typedef int          i32x8 __attribute__((ext_vector_type(8)));
typedef int          i32x4 __attribute__((ext_vector_type(4)));

#define WMMA_BF16(a, b, c) \
  __builtin_amdgcn_wmma_f32_16x16x32_bf16(false, (a), false, (b), (short)0, (c), false, false)

static __device__ __forceinline__ v8f zero8() {
  v8f z = {0.f, 0.f, 0.f, 0.f, 0.f, 0.f, 0.f, 0.f};
  return z;
}

// Shapes
#define Bn 8
#define Tn 2048
#define En 1024
#define Hn 64
#define SCALE 0.03125f   // E^-0.5 = 1024^-0.5

// ---------------------------------------------------------------------------
// TDM: 1-D bulk copy global -> LDS, nbytes must be a multiple of 8.
// D# per CDNA5 ISA ch.8: group0 = {count=1, lds_addr, global_addr, type=2},
// group1 packs data_size=8B, tensor_dim0 = tile_dim0 = n (8B units),
// tensor_dim0_stride = n. Tracked with TENSORcnt; one issue per wave.
// ---------------------------------------------------------------------------
static __device__ __forceinline__ void tdm_load_1d(unsigned lds_off,
                                                   const void* gptr,
                                                   unsigned nbytes) {
  unsigned long long ga = (unsigned long long)(uintptr_t)gptr;
  unsigned n = nbytes >> 3;                       // 8-byte elements
  u32x4 g0;
  g0[0] = 1u;                                     // count=1, user descriptor
  g0[1] = lds_off;                                // lds_addr
  g0[2] = (unsigned)(ga & 0xFFFFFFFFu);           // global_addr[31:0]
  g0[3] = (unsigned)((ga >> 32) & 0x01FFFFFFu) | (2u << 30);  // addr[56:32] | type=2
  i32x8 g1;
  g1[0] = (int)(3u << 16);                        // data_size=3 (8B), mask=0, no flags
  g1[1] = (int)((n & 0xFFFFu) << 16);             // tensor_dim0[15:0] @ bits63:48
  g1[2] = (int)(((n >> 16) & 0xFFFFu) | (1u << 16)); // dim0 hi | tensor_dim1=1
  g1[3] = (int)((n & 0xFFFFu) << 16);             // tile_dim0 @ bits127:112
  g1[4] = 0;                                      // tile_dim1/2 unused
  g1[5] = (int)n;                                 // tensor_dim0_stride[31:0]
  g1[6] = 0;
  g1[7] = 0;
  i32x4 z4 = {0, 0, 0, 0};
#if __clang_major__ >= 23
  i32x8 z8 = {0, 0, 0, 0, 0, 0, 0, 0};
  __builtin_amdgcn_tensor_load_to_lds(g0, g1, z4, z4, z8, 0);
#else
  __builtin_amdgcn_tensor_load_to_lds(g0, g1, z4, z4, 0);
#endif
}

// ---------------------------------------------------------------------------
// Kernel 0: repack Wq/Wk/Wv (f32 [E,H]) into bf16 B-matrix WMMA lane layout.
// Chunk-major tile index = (c*3 + m)*4 + nt so one E-chunk (12 tiles, 12KB)
// is a single contiguous run for the TDM. B layout (32x16, KxN): lane n<16
// holds col N=n, elems j -> K=j; lane n+16 holds col N=n, elems j -> K=16+j.
// ---------------------------------------------------------------------------
__global__ void pack_weights(const float* __restrict__ Wq,
                             const float* __restrict__ Wk,
                             const float* __restrict__ Wv,
                             __bf16* __restrict__ wB) {
  int tile = blockIdx.x;            // 0..383
  int l    = threadIdx.x;           // 0..31
  int c    = tile / 12;             // E-chunk 0..31
  int rem  = tile - c * 12;
  int m    = rem >> 2;              // 0=q 1=k 2=v
  int nt   = rem & 3;               // H/16 tile
  const float* W = (m == 0) ? Wq : ((m == 1) ? Wk : Wv);
  int col  = nt * 16 + (l & 15);
  int koff = (l < 16) ? 0 : 16;
  int e0   = c * 32;
  v16bf val;
#pragma unroll
  for (int j = 0; j < 16; ++j)
    val[j] = (__bf16)W[(size_t)(e0 + koff + j) * Hn + col];
  *(v16bf*)(wB + (size_t)tile * 512 + l * 16) = val;
}

// ---------------------------------------------------------------------------
// Kernel 1: fused Q/K/V projection + repack into WMMA-native global layouts.
// Weight chunks (12KB) are TDM-staged through a double-buffered LDS stage,
// shared by all 8 waves; x tiles stream per-lane from global with prefetch.
// ---------------------------------------------------------------------------
static __device__ __forceinline__ v16bf load_a_tile(const float* xrow, int c,
                                                    int kbaseA) {
  const float4* xp0 = (const float4*)(xrow + c * 32 + 2 * kbaseA);
  const float4* xp1 = (const float4*)(xrow + c * 32 + 2 * kbaseA + 8);
  float4 f0 = xp0[0], f1 = xp0[1];
  float4 f2 = xp1[0], f3 = xp1[1];
  v16bf a;
  a[0] = (__bf16)f0.x;  a[1] = (__bf16)f0.y;  a[2] = (__bf16)f0.z;  a[3] = (__bf16)f0.w;
  a[4] = (__bf16)f1.x;  a[5] = (__bf16)f1.y;  a[6] = (__bf16)f1.z;  a[7] = (__bf16)f1.w;
  a[8] = (__bf16)f2.x;  a[9] = (__bf16)f2.y;  a[10] = (__bf16)f2.z; a[11] = (__bf16)f2.w;
  a[12] = (__bf16)f3.x; a[13] = (__bf16)f3.y; a[14] = (__bf16)f3.z; a[15] = (__bf16)f3.w;
  return a;
}

__global__ __launch_bounds__(256)
void proj_kernel(const float* __restrict__ x, const __bf16* __restrict__ wB,
                 __bf16* __restrict__ qA, __bf16* __restrict__ kB,
                 __bf16* __restrict__ vB) {
  __shared__ __bf16 lq[128 * 64];
  __shared__ __bf16 lk[128 * 64];
  __shared__ __bf16 lv[128 * 64];
  __shared__ __bf16 wbuf[2][6144];   // double-buffered 12KB weight chunk stage

  int b      = blockIdx.x >> 4;
  int tblk   = blockIdx.x & 15;
  int w      = threadIdx.x >> 5;
  int l      = threadIdx.x & 31;
  int half8  = (l < 16) ? 0 : 8;
  int row16  = l & 15;
  int kbaseA = (l < 16) ? 0 : 8;

  v8f aq[4], ak[4], av[4];
#pragma unroll
  for (int nt = 0; nt < 4; ++nt) { aq[nt] = zero8(); ak[nt] = zero8(); av[nt] = zero8(); }

  const float* xrow = x + ((size_t)b * Tn + tblk * 128 + w * 16 + row16) * En;

  unsigned wb0 = (unsigned)(uintptr_t)&wbuf[0][0];
  unsigned wb1 = (unsigned)(uintptr_t)&wbuf[1][0];
  if (w == 0) tdm_load_1d(wb0, wB, 12288);   // chunk 0

  v16bf a = load_a_tile(xrow, 0, kbaseA);
  for (int c = 0; c < 32; ++c) {
    __syncthreads();                 // buffer (c+1)&1 free for DMA
    if (w == 0) {
      if (c < 31) {
        tdm_load_1d(((c + 1) & 1) ? wb1 : wb0, wB + (size_t)(c + 1) * 6144, 12288);
        __builtin_amdgcn_s_wait_tensorcnt(1);   // chunk c's DMA done
      } else {
        __builtin_amdgcn_s_wait_tensorcnt(0);
      }
    }
    __syncthreads();                 // buffer c&1 valid for all waves

    const __bf16* wc = &wbuf[c & 1][0] + l * 16;
    __builtin_prefetch(xrow + (c + 4) * 32, 0, 1);
    v16bf anext = a;
    if (c < 31) anext = load_a_tile(xrow, c + 1, kbaseA);

    // 12 WMMAs, B tiles from LDS (tile local index = m*4+nt, 512 bf16 apart)
    aq[0] = WMMA_BF16(a, *(const v16bf*)(wc + 0 * 512), aq[0]);
    aq[1] = WMMA_BF16(a, *(const v16bf*)(wc + 1 * 512), aq[1]);
    aq[2] = WMMA_BF16(a, *(const v16bf*)(wc + 2 * 512), aq[2]);
    aq[3] = WMMA_BF16(a, *(const v16bf*)(wc + 3 * 512), aq[3]);
    ak[0] = WMMA_BF16(a, *(const v16bf*)(wc + 4 * 512), ak[0]);
    ak[1] = WMMA_BF16(a, *(const v16bf*)(wc + 5 * 512), ak[1]);
    ak[2] = WMMA_BF16(a, *(const v16bf*)(wc + 6 * 512), ak[2]);
    ak[3] = WMMA_BF16(a, *(const v16bf*)(wc + 7 * 512), ak[3]);
    av[0] = WMMA_BF16(a, *(const v16bf*)(wc + 8 * 512), av[0]);
    av[1] = WMMA_BF16(a, *(const v16bf*)(wc + 9 * 512), av[1]);
    av[2] = WMMA_BF16(a, *(const v16bf*)(wc + 10 * 512), av[2]);
    av[3] = WMMA_BF16(a, *(const v16bf*)(wc + 11 * 512), av[3]);
    a = anext;
  }

  // Spill C-layout accumulators to LDS row-major bf16 [128][64].
#pragma unroll
  for (int nt = 0; nt < 4; ++nt) {
#pragma unroll
    for (int v = 0; v < 8; ++v) {
      int row = w * 16 + v + half8;
      int col = nt * 16 + row16;
      lq[row * 64 + col] = (__bf16)aq[nt][v];
      lk[row * 64 + col] = (__bf16)ak[nt][v];
      lv[row * 64 + col] = (__bf16)av[nt][v];
    }
  }
  __syncthreads();

  // --- pack Q into A-layout tiles ---
#pragma unroll
  for (int hc = 0; hc < 2; ++hc) {
    v16bf qa;
#pragma unroll
    for (int j = 0; j < 16; ++j) {
      int K = 2 * kbaseA + j;
      qa[j] = lq[(w * 16 + row16) * 64 + hc * 32 + K];
    }
    size_t tile = ((size_t)(b * 128 + tblk * 8 + w) * 2 + hc);
    *(v16bf*)(qA + tile * 512 + l * 16) = qa;
  }

  // --- pack K^T into B-layout tiles (32h x 16keys) ---
  int koff16 = (l < 16) ? 0 : 16;
#pragma unroll
  for (int hc = 0; hc < 2; ++hc) {
    v16bf kb = *(const v16bf*)&lk[(w * 16 + row16) * 64 + hc * 32 + koff16];
    size_t tile = ((size_t)(b * 128 + tblk * 8 + w) * 2 + hc);
    *(v16bf*)(kB + tile * 512 + l * 16) = kb;
  }

  // --- pack V into B-layout tiles (32keys x 16h) ---
#pragma unroll
  for (int i = 0; i < 2; ++i) {
    int idx = w * 2 + i;
    int kcl = idx >> 2;
    int ht  = idx & 3;
    v16bf vb;
#pragma unroll
    for (int j = 0; j < 16; ++j) {
      int K = (l < 16) ? j : (16 + j);
      vb[j] = lv[(kcl * 32 + K) * 64 + ht * 16 + row16];
    }
    size_t tile = ((size_t)(b * 64 + tblk * 4 + kcl) * 4 + ht);
    *(v16bf*)(vB + tile * 512 + l * 16) = vb;
  }
}

// ---------------------------------------------------------------------------
// Kernel 2: flash attention with TDM-staged K/V chunks.
// Block = 8 waves, each owns a 16-query tile. Per 32-key chunk, wave 0 DMAs
// the 4KB K^T-tile run + 4KB V-tile run into a double-buffered LDS stage,
// then a barrier publishes it to all waves. All waves iterate to the block's
// max causal chunk (uniform barriers); compute is predicated per wave.
// ---------------------------------------------------------------------------
__global__ __launch_bounds__(256)
void attn_kernel(const __bf16* __restrict__ qA, const __bf16* __restrict__ kB,
                 const __bf16* __restrict__ vB, float* __restrict__ out) {
  __shared__ __bf16 skv[2][4096];       // per buf: 2048 bf16 K-run + 2048 bf16 V-run
  __shared__ __bf16 pbuf[8][16 * 32];   // per-wave P transpose scratch

  int b     = blockIdx.x >> 4;
  int tblk  = blockIdx.x & 15;
  int w     = threadIdx.x >> 5;
  int l     = threadIdx.x & 31;
  int half8 = (l < 16) ? 0 : 8;
  int n16   = l & 15;
  int qt    = tblk * 8 + w;
  int q0    = qt * 16;

  v16bf qreg[2];
#pragma unroll
  for (int hc = 0; hc < 2; ++hc)
    qreg[hc] = *(const v16bf*)(qA + ((size_t)(b * 128 + qt) * 2 + hc) * 512 + l * 16);

  v8f o[4];
#pragma unroll
  for (int ht = 0; ht < 4; ++ht) o[ht] = zero8();
  float mrow[8], lrow[8];
#pragma unroll
  for (int v = 0; v < 8; ++v) { mrow[v] = -1e30f; lrow[v] = 0.f; }

  __bf16* pw = &pbuf[w][0];
  int kcEndW   = (q0 + 15) >> 5;        // this wave's last causal chunk
  int kcEndBlk = tblk * 4 + 3;          // block's last chunk (wave 7's)

  unsigned lds_k0 = (unsigned)(uintptr_t)&skv[0][0];
  unsigned lds_k1 = (unsigned)(uintptr_t)&skv[1][0];

  if (w == 0) {
    tdm_load_1d(lds_k0,        kB + ((size_t)(b * 128) * 2) * 512, 4096);
    tdm_load_1d(lds_k0 + 4096, vB + ((size_t)(b * 64) * 4) * 512,  4096);
  }

  for (int kc = 0; kc <= kcEndBlk; ++kc) {
    __syncthreads();                    // buffer (kc+1)&1 free for DMA
    if (w == 0) {
      if (kc < kcEndBlk) {
        int kn = kc + 1;
        unsigned dst = (kn & 1) ? lds_k1 : lds_k0;
        tdm_load_1d(dst,        kB + ((size_t)(b * 128 + kn * 2) * 2) * 512, 4096);
        tdm_load_1d(dst + 4096, vB + ((size_t)(b * 64 + kn) * 4) * 512,      4096);
        __builtin_amdgcn_s_wait_tensorcnt(2);   // chunk kc's 2 DMAs done
      } else {
        __builtin_amdgcn_s_wait_tensorcnt(0);
      }
    }
    __syncthreads();                    // buffer kc&1 valid for all waves

    if (kc <= kcEndW) {
      const __bf16* kbase = &skv[kc & 1][0];
      const __bf16* vbase = &skv[kc & 1][2048];

      // S = Q * K^T  (two 16x16 key tiles, K-dim = H = 64 via 2 WMMAs each)
      v8f s[2];
#pragma unroll
      for (int kt = 0; kt < 2; ++kt) {
        v16bf k0 = *(const v16bf*)(kbase + (kt * 2 + 0) * 512 + l * 16);
        v16bf k1 = *(const v16bf*)(kbase + (kt * 2 + 1) * 512 + l * 16);
        s[kt] = WMMA_BF16(qreg[0], k0, zero8());
        s[kt] = WMMA_BF16(qreg[1], k1, s[kt]);
      }

      // scale + causal mask
#pragma unroll
      for (int kt = 0; kt < 2; ++kt) {
        int key = (kc * 2 + kt) * 16 + n16;
#pragma unroll
        for (int v = 0; v < 8; ++v) {
          float val = s[kt][v] * SCALE;
          int row = q0 + v + half8;
          s[kt][v] = (key <= row) ? val : -1e30f;
        }
      }

      // online softmax; each row lives in one 16-lane half
      float alpha[8];
#pragma unroll
      for (int v = 0; v < 8; ++v) {
        float cm = fmaxf(s[0][v], s[1][v]);
        cm = fmaxf(cm, __shfl_xor(cm, 1, 16));
        cm = fmaxf(cm, __shfl_xor(cm, 2, 16));
        cm = fmaxf(cm, __shfl_xor(cm, 4, 16));
        cm = fmaxf(cm, __shfl_xor(cm, 8, 16));
        float mnew = fmaxf(mrow[v], cm);
        alpha[v] = __expf(mrow[v] - mnew);
        float p0 = __expf(s[0][v] - mnew);
        float p1 = __expf(s[1][v] - mnew);
        s[0][v] = p0; s[1][v] = p1;
        float rs = p0 + p1;
        rs += __shfl_xor(rs, 1, 16);
        rs += __shfl_xor(rs, 2, 16);
        rs += __shfl_xor(rs, 4, 16);
        rs += __shfl_xor(rs, 8, 16);
        lrow[v] = lrow[v] * alpha[v] + rs;
        mrow[v] = mnew;
      }
#pragma unroll
      for (int ht = 0; ht < 4; ++ht)
#pragma unroll
        for (int v = 0; v < 8; ++v) o[ht][v] *= alpha[v];

      // P: C-layout -> A-layout via wave-private LDS scratch
#pragma unroll
      for (int kt = 0; kt < 2; ++kt)
#pragma unroll
        for (int v = 0; v < 8; ++v)
          pw[(v + half8) * 32 + kt * 16 + n16] = (__bf16)s[kt][v];

      v8bf lo = *(const v8bf*)&pw[n16 * 32 + 2 * half8];
      v8bf hi = *(const v8bf*)&pw[n16 * 32 + 2 * half8 + 8];
      v16bf pa;
#pragma unroll
      for (int j = 0; j < 8; ++j) { pa[j] = lo[j]; pa[8 + j] = hi[j]; }

      // O += P * V
#pragma unroll
      for (int ht = 0; ht < 4; ++ht) {
        v16bf vb = *(const v16bf*)(vbase + ht * 512 + l * 16);
        o[ht] = WMMA_BF16(pa, vb, o[ht]);
      }
    }
  }

  // epilogue
#pragma unroll
  for (int v = 0; v < 8; ++v) {
    float inv = 1.0f / lrow[v];
    int row = q0 + v + half8;
#pragma unroll
    for (int ht = 0; ht < 4; ++ht)
      out[((size_t)b * Tn + row) * Hn + ht * 16 + n16] = o[ht][v] * inv;
  }
}

// ---------------------------------------------------------------------------
extern "C" void kernel_launch(void* const* d_in, const int* in_sizes, int n_in,
                              void* d_out, int out_size, void* d_ws, size_t ws_size,
                              hipStream_t stream) {
  const float* x  = (const float*)d_in[0];
  const float* Wk = (const float*)d_in[1];
  const float* Wq = (const float*)d_in[2];
  const float* Wv = (const float*)d_in[3];

  char* ws = (char*)d_ws;
  __bf16* qA  = (__bf16*)(ws);                          // 2 MB
  __bf16* kBp = (__bf16*)(ws + (size_t)(2u << 20));     // 2 MB
  __bf16* vBp = (__bf16*)(ws + (size_t)(4u << 20));     // 2 MB
  __bf16* wB  = (__bf16*)(ws + (size_t)(6u << 20));     // 384 KB

  pack_weights<<<384, 32, 0, stream>>>(Wq, Wk, Wv, wB);
  proj_kernel<<<128, 256, 0, stream>>>(x, wB, qA, kBp, vBp);
  attn_kernel<<<128, 256, 0, stream>>>(qA, kBp, vBp, (float*)d_out);
}
